// LPLLoss_21998822490239
// MI455X (gfx1250) — compile-verified, matrix operand order checked by reading, not run on hardware
//
#include <hip/hip_runtime.h>
#include <hip/hip_bf16.h>
#include <math.h>

typedef __attribute__((ext_vector_type(16))) _Float16       v16h;
typedef __attribute__((ext_vector_type(8)))  _Float16       v8h;
typedef __attribute__((ext_vector_type(8)))  unsigned short v8us;
typedef __attribute__((ext_vector_type(8)))  float          v8f;

#define NCLS   10
#define PGD    10
#define ALPHA  0.15f

// ---- workspace layout (in floats) ----
#define WS_S     0                        // [PGD][256]  per-step 16x16 f32 class sums
#define WS_CNT   (PGD * 256)              // [16]        per-class counts (as float)
#define WS_CUM   (WS_CNT + 16)            // [PGD+1][100] cumulative per-class deltas
#define WS_TOTAL (WS_CUM + (PGD + 1) * 100)

// -------------------------------------------------------------------------
// Zero all scratch + the loss accumulator cell (graph replay safe).
__global__ __launch_bounds__(256) void lpl_init(float* __restrict__ ws,
                                                float* __restrict__ out) {
  int tid = blockIdx.x * blockDim.x + threadIdx.x;
  for (int i = tid; i < WS_TOTAL; i += gridDim.x * blockDim.x) ws[i] = 0.0f;
  if (tid == 0) out[0] = 0.0f;
}

// -------------------------------------------------------------------------
// Per-class counts (constant across steps): LDS histogram, then 16 atomics.
__global__ __launch_bounds__(256) void lpl_counts(const int* __restrict__ y,
                                                  float* __restrict__ cnt, int B) {
  __shared__ float h[16];
  int tid = threadIdx.x;
  if (tid < 16) h[tid] = 0.0f;
  __syncthreads();
  int stride = gridDim.x * blockDim.x;
  for (int n = blockIdx.x * blockDim.x + tid; n < B; n += stride)
    atomicAdd(&h[y[n]], 1.0f);
  __syncthreads();
  if (tid < 16) atomicAdd(&cnt[tid], h[tid]);
}

// -------------------------------------------------------------------------
// One PGD step: S_t[c][j] = sum over {n : y[n]==c} of softmax(logit0[n]+cum_t[c])[j]
// Segment-sum realized as onehot^T @ P with v_wmma_f32_16x16x32_f16 (32 samples/WMMA).
// LDS staging is transposed so the B-operand column is 2x ds_load_b128 per lane.
__global__ __launch_bounds__(256) void lpl_step(const float* __restrict__ logit,
                                                const int*   __restrict__ y,
                                                const float* __restrict__ cum_t,
                                                float*       __restrict__ S_out,
                                                int B) {
  __shared__ __attribute__((aligned(16))) _Float16       s_pT[8][16][32]; // [wave][class][sample]
  __shared__ __attribute__((aligned(16))) unsigned short s_y16[8][32];
  __shared__ float s_cum[NCLS * NCLS];
  __shared__ float s_S[256];

  const int tid  = threadIdx.x;
  const int w    = tid >> 5;
  const int lane = tid & 31;

  if (tid < NCLS * NCLS) s_cum[tid] = cum_t[tid];
  s_S[tid] = 0.0f;
  // padded class rows 10..15 are never written again: zero once (wave-private)
  #pragma unroll
  for (int j = NCLS; j < 16; ++j) s_pT[w][j][lane] = (_Float16)0.0f;
  __syncthreads();

  const int ntiles     = (B + 31) >> 5;
  const int totalWaves = (gridDim.x * blockDim.x) >> 5;
  const int gw         = (blockIdx.x * blockDim.x + tid) >> 5;
  const int iters      = (ntiles + totalWaves - 1) / totalWaves;

  const int m  = lane & 15;
  const int kb = (lane < 16) ? 0 : 8;   // A/B 16-bit K split per ISA layout tables

  v8f acc = {};

  for (int it = 0; it < iters; ++it) {
    int  tile   = gw + it * totalWaves;
    bool tvalid = tile < ntiles;                     // wave-uniform
    if (tvalid) {
      int   n = (tile << 5) + lane;
      float p[NCLS];
      unsigned short c16 = 0xFFFFu;                  // invalid row -> matches no class
      if (n < B) {
        int c = y[n];
        c16 = (unsigned short)c;
        const float2* r2 = (const float2*)(logit + (size_t)n * NCLS); // 8B aligned
        float l[NCLS];
        #pragma unroll
        for (int j = 0; j < 5; ++j) { float2 v = r2[j]; l[2*j] = v.x; l[2*j+1] = v.y; }
        #pragma unroll
        for (int j = 0; j < NCLS; ++j) l[j] += s_cum[c * NCLS + j];
        float mx = l[0];
        #pragma unroll
        for (int j = 1; j < NCLS; ++j) mx = fmaxf(mx, l[j]);
        float s = 0.0f;
        #pragma unroll
        for (int j = 0; j < NCLS; ++j) { p[j] = __expf(l[j] - mx); s += p[j]; }
        float inv = 1.0f / s;
        #pragma unroll
        for (int j = 0; j < NCLS; ++j) p[j] *= inv;
      } else {
        #pragma unroll
        for (int j = 0; j < NCLS; ++j) p[j] = 0.0f;
      }
      #pragma unroll
      for (int j = 0; j < NCLS; ++j) s_pT[w][j][lane] = (_Float16)p[j];
      s_y16[w][lane] = c16;

      __builtin_amdgcn_wave_barrier();   // wave-private LDS: per-wave DS in-order

      // B operand: column m of the softmax tile, K chunks [kb..kb+7], [kb+16..kb+23]
      v8h blo = *(const v8h*)&s_pT[w][m][kb];
      v8h bhi = *(const v8h*)&s_pT[w][m][kb + 16];
      v16h b  = __builtin_shufflevector(blo, bhi, 0,1,2,3,4,5,6,7,8,9,10,11,12,13,14,15);
      // A operand: one-hot row m over the same K chunks
      v8us ylo = *(const v8us*)&s_y16[w][kb];
      v8us yhi = *(const v8us*)&s_y16[w][kb + 16];
      v16h a;
      #pragma unroll
      for (int h = 0; h < 8; ++h) {
        a[h]     = (ylo[h] == (unsigned short)m) ? (_Float16)1.0f : (_Float16)0.0f;
        a[h + 8] = (yhi[h] == (unsigned short)m) ? (_Float16)1.0f : (_Float16)0.0f;
      }
      acc = __builtin_amdgcn_wmma_f32_16x16x32_f16(false, a, false, b,
                                                   (short)0, acc, false, false);
      __builtin_amdgcn_wave_barrier();   // keep next-iter stores after this-iter reads
    }
  }

  // C/D layout: VGPR v of lane L holds [M = v + 8*(L>=16), N = L&15]
  #pragma unroll
  for (int v = 0; v < 8; ++v) {
    int mm = v + ((lane >= 16) ? 8 : 0);
    atomicAdd(&s_S[mm * 16 + (lane & 15)], acc[v]);
  }
  __syncthreads();
  atomicAdd(&S_out[tid], s_S[tid]);
}

// -------------------------------------------------------------------------
// Tiny 10x10 solve: grad = S/cnt - I; row c scaled by 1/||col_c||, ALPHA, sign[c].
__global__ __launch_bounds__(128) void lpl_update(const float* __restrict__ S,
                                                  const float* __restrict__ cnt,
                                                  const float* __restrict__ cum_in,
                                                  float*       __restrict__ cum_out) {
  __shared__ float g[NCLS][NCLS];
  int tid = threadIdx.x;
  int c = tid / NCLS, j = tid % NCLS;
  if (tid < NCLS * NCLS) {
    float k = cnt[c];
    if (k == 0.0f) k = 100.0f;
    g[c][j] = S[c * 16 + j] / k - ((c == j) ? 1.0f : 0.0f);
  }
  __syncthreads();
  if (tid < NCLS * NCLS) {
    float cn = 0.0f;
    #pragma unroll
    for (int i = 0; i < NCLS; ++i) cn += g[i][c] * g[i][c];   // column-c norm
    cn = sqrtf(cn);
    float sign = (c < 4) ? -1.0f : 1.0f;
    cum_out[tid] = cum_in[tid] + ALPHA * sign * g[c][j] / cn;
  }
}

// -------------------------------------------------------------------------
// Fused gather + outputs + mean cross-entropy.
__global__ __launch_bounds__(256) void lpl_final(const float* __restrict__ logit,
                                                 const int*   __restrict__ y,
                                                 const int*   __restrict__ lf,
                                                 const float* __restrict__ cum_steps,
                                                 float*       __restrict__ out,
                                                 int B) {
  __shared__ float s_cum[(PGD + 1) * NCLS * NCLS];
  __shared__ int   s_lf[NCLS];
  int tid = threadIdx.x;
  for (int i = tid; i < (PGD + 1) * NCLS * NCLS; i += blockDim.x) s_cum[i] = cum_steps[i];
  if (tid < NCLS) s_lf[tid] = lf[tid];
  __syncthreads();

  float* loss = out;
  float* news = out + 1;                         // 4B aligned only -> scalar stores
  float* copy = out + 1 + (size_t)B * NCLS;
  const float invB = 1.0f / (float)B;

  int idx    = blockIdx.x * blockDim.x + tid;
  int stride = gridDim.x * blockDim.x;
  int iters  = (B + stride - 1) / stride;        // uniform -> shuffles stay converged
  int lane   = tid & 31;

  for (int it = 0; it < iters; ++it) {
    int   n = idx + it * stride;
    float contrib = 0.0f;
    if (n < B) {
      int c = y[n];
      int k = s_lf[c] + 1;                       // logit_steps[s] is after s+1 updates
      const float* cm = &s_cum[(k * NCLS + c) * NCLS];
      const float2* r2 = (const float2*)(logit + (size_t)n * NCLS);
      float r[NCLS];
      #pragma unroll
      for (int j = 0; j < 5; ++j) {
        float2 v = r2[j];
        copy[(size_t)n * NCLS + 2 * j]     = v.x;
        copy[(size_t)n * NCLS + 2 * j + 1] = v.y;
        r[2 * j]     = v.x + cm[2 * j];
        r[2 * j + 1] = v.y + cm[2 * j + 1];
      }
      #pragma unroll
      for (int j = 0; j < NCLS; ++j) news[(size_t)n * NCLS + j] = r[j];
      float m = r[0];
      #pragma unroll
      for (int j = 1; j < NCLS; ++j) m = fmaxf(m, r[j]);
      float s = 0.0f;
      #pragma unroll
      for (int j = 0; j < NCLS; ++j) s += __expf(r[j] - m);
      contrib = -(r[c] - m - __logf(s)) * invB;
    }
    #pragma unroll
    for (int off = 16; off > 0; off >>= 1)
      contrib += __shfl_xor(contrib, off, 32);
    if (lane == 0) atomicAdd(loss, contrib);
  }
}

// -------------------------------------------------------------------------
extern "C" void kernel_launch(void* const* d_in, const int* in_sizes, int n_in,
                              void* d_out, int out_size, void* d_ws, size_t ws_size,
                              hipStream_t stream) {
  const float* logit = (const float*)d_in[0];
  const int*   y     = (const int*)d_in[1];
  const int*   lf    = (const int*)d_in[2];
  float*       out   = (float*)d_out;
  float*       ws    = (float*)d_ws;
  const int    B     = in_sizes[1];

  float* S   = ws + WS_S;
  float* cnt = ws + WS_CNT;
  float* cum = ws + WS_CUM;   // cum[0] == 0 after init

  lpl_init  <<<1,   256, 0, stream>>>(ws, out);
  lpl_counts<<<256, 256, 0, stream>>>(y, cnt, B);
  for (int t = 0; t < PGD; ++t) {
    // 512 blocks * 8 waves = 4096 waves; 32768 tiles -> exactly 8 tiles/wave
    lpl_step  <<<512, 256, 0, stream>>>(logit, y, cum + t * 100, S + t * 256, B);
    lpl_update<<<1,   128, 0, stream>>>(S + t * 256, cnt, cum + t * 100,
                                        cum + (t + 1) * 100);
  }
  lpl_final<<<1024, 256, 0, stream>>>(logit, y, lf, cum, out, B);
}
